// MultiHeadAttention_18193481466297
// MI455X (gfx1250) — compile-verified, matrix operand order checked by reading, not run on hardware
//
#include <hip/hip_runtime.h>
#include <hip/hip_bf16.h>
#include <hip/hip_fp16.h>

typedef __attribute__((ext_vector_type(16))) _Float16 v16h;
typedef __attribute__((ext_vector_type(8)))  float    v8f;
typedef __attribute__((ext_vector_type(4)))  unsigned u32x4;
typedef __attribute__((ext_vector_type(8)))  int      i32x8;
typedef __attribute__((ext_vector_type(4)))  int      i32x4;

#define B_  4
#define S_  2048
#define D_  768
#define H_  12
#define E_  64
#define HE_ 768
#define LN_EPS 1e-5f
#define ISCALE 0.02209708691f   // 1/sqrt(2048) -- reference scales by sqrt(seq_len)

#if defined(__has_builtin)
#if __has_builtin(__builtin_amdgcn_tensor_load_to_lds)
#define HAVE_TDM 1
#endif
#endif
#ifndef HAVE_TDM
#define HAVE_TDM 0
#endif

// ---------------------------------------------------------------------------
// WMMA fragment helpers (wave32, V_WMMA_F32_16X16X32_F16)
//
// 16-bit A (16xK) / B(col-major) fragment layout per ISA 7.12.2:
//   lane l: i = l&15 (row for A / col for B), hi = l>>4
//   VGPR v: k = kbase + (v>>2)*16 + hi*8 + (v&3)*2   (pair k, k+1)
// Requires element (i,k) at p[i*ld + k] with contiguous k-pairs.
// ---------------------------------------------------------------------------
__device__ __forceinline__ v16h frag_ld(const _Float16* p, int ld, int kbase) {
  const int l  = threadIdx.x & 31;
  const int i  = l & 15;
  const int hi = l >> 4;
  union { v16h h; unsigned u[8]; } f;
#pragma unroll
  for (int v = 0; v < 8; ++v) {
    int k = kbase + ((v >> 2) << 4) + (hi << 3) + ((v & 3) << 1);
    f.u[v] = *(const unsigned*)(p + i * ld + k);
  }
  return f.h;
}

__device__ __forceinline__ v8f wmma16(v16h a, v16h b, v8f c) {
  return __builtin_amdgcn_wmma_f32_16x16x32_f16(false, a, false, b, (short)0, c,
                                                false, false);
}

// ---------------------------------------------------------------------------
// f32 -> f16 elementwise cast
// ---------------------------------------------------------------------------
__global__ void cast_f16_kernel(const float* __restrict__ in,
                                _Float16* __restrict__ out, int n) {
  int i = blockIdx.x * blockDim.x + threadIdx.x;
  if (i < n) out[i] = (_Float16)in[i];
}

// Wo [rows=HE, cols=D] f32 -> WoT [cols][rows] f16 (K-pairs contiguous for B frags)
__global__ void packT_kernel(const float* __restrict__ in,
                             _Float16* __restrict__ out, int rows, int cols) {
  int c = blockIdx.x * 16 + threadIdx.x;
  int r = blockIdx.y * 16 + threadIdx.y;
  if (r < rows && c < cols)
    out[(size_t)c * rows + r] = (_Float16)in[(size_t)r * cols + c];
}

// ---------------------------------------------------------------------------
// QKV projection: per (b,h):  X[S,D] @ W[D,E] + b  -> Q/K [b,h,s,e], V^T [b,h,e,s]
// Block: 128 threads (4 waves), 64x64 tile.
// X tile staged to LDS by the Tensor Data Mover (D# descriptor, LDS padding
// reproduces the 36-half conflict-avoidance stride); W^T tile staged by lanes.
// ---------------------------------------------------------------------------
__global__ __launch_bounds__(128) void qkv_kernel(
    const _Float16* __restrict__ xh,
    const _Float16* __restrict__ Wq, const _Float16* __restrict__ Wk,
    const _Float16* __restrict__ Wv,
    const float* __restrict__ bq, const float* __restrict__ bk,
    const float* __restrict__ bv,
    _Float16* __restrict__ Qo, _Float16* __restrict__ Ko,
    _Float16* __restrict__ Vo) {
  __shared__ __align__(16) _Float16 xs[64 * 36];   // X tile   [row][k], pad 36
  __shared__ __align__(16) _Float16 ws[64 * 36];   // W^T tile [n][k],  pad 36

  const int z  = blockIdx.z;                       // 0=Q 1=K 2=V
  const int bh = blockIdx.y;
  const int b  = bh / H_, h = bh % H_;
  const int m0 = blockIdx.x * 64;

  const _Float16* W    = (z == 0) ? Wq : (z == 1) ? Wk : Wv;
  const float*    bias = (z == 0) ? bq : (z == 1) ? bk : bv;
  _Float16*       outp = (z == 0) ? Qo : (z == 1) ? Ko : Vo;

  const int t    = threadIdx.x;
  const int wid  = t >> 5;
  const int lane = t & 31;

  v8f acc[4];
#pragma unroll
  for (int i = 0; i < 4; ++i) {
    v8f zv = {0.f, 0.f, 0.f, 0.f, 0.f, 0.f, 0.f, 0.f};
    acc[i] = zv;
  }

  for (int kb = 0; kb < D_; kb += 32) {
#if HAVE_TDM
    // --- X tile via Tensor Data Mover: 64 rows x 32 halves -> LDS ---------
    if (wid == 0) {
      const unsigned lds_base =
          (unsigned)(unsigned long long)(uintptr_t)&xs[0];   // generic->LDS offset
      const unsigned long long ga =
          (unsigned long long)(uintptr_t)(xh + ((size_t)b * S_ + m0) * D_ + kb);
      u32x4 g0;
      g0[0] = 1u;                                   // count=1 valid descriptor
      g0[1] = lds_base;                             // lds_addr (bytes)
      g0[2] = (unsigned)(ga & 0xFFFFFFFFu);         // global_addr[31:0]
      g0[3] = (unsigned)((ga >> 32) & 0x01FFFFFFu)  // global_addr[56:32]
              | (2u << 30);                         // type = 2 ("image")
      i32x8 g1;
      g1[0] = (1 << 16)      // data_size = 2 bytes
            | (1 << 20)      // pad_enable
            | (3 << 22)      // pad_interval: every 16 DWORDs (one 64B row)
            | (1 << 25);     // pad_amount: 2 DWORDs (-> 36-half LDS stride)
      g1[1] = (int)(((unsigned)D_ & 0xFFFFu) << 16);          // tensor_dim0 lo16
      g1[2] = (int)((((unsigned)(B_ * S_)) & 0xFFFFu) << 16); // dim0 hi=0, dim1 lo16
      g1[3] = (int)(32u << 16);                               // dim1 hi=0, tile_dim0=32
      g1[4] = 64;                                             // tile_dim1=64, tile_dim2=0
      g1[5] = D_;                                             // tensor_dim0_stride lo32
      g1[6] = 0;                                              // stride hi, dim1_stride lo
      g1[7] = 0;
      i32x4 gz = {0, 0, 0, 0};
#if __clang_major__ >= 23
      i32x8 gz8 = {0, 0, 0, 0, 0, 0, 0, 0};
      __builtin_amdgcn_tensor_load_to_lds(g0, g1, gz, gz, gz8, 0);
#else
      __builtin_amdgcn_tensor_load_to_lds(g0, g1, gz, gz, 0);
#endif
    }
#else
    // --- X tile staged by lanes: 2 threads/row, 8 dwords each -------------
    {
      const int rr = t >> 1, hf = t & 1;
      const _Float16* gx = xh + ((size_t)b * S_ + m0 + rr) * D_ + kb + hf * 16;
      _Float16*       lx = xs + rr * 36 + hf * 16;
#pragma unroll
      for (int j = 0; j < 8; ++j)
        *(unsigned*)(lx + j * 2) = *(const unsigned*)(gx + j * 2);
    }
#endif
    // --- W tile transposed: 32 k x 64 n -> ws[n][k] -----------------------
    {
      const int kk = t >> 2, g = t & 3;
      const _Float16* gw = W + (size_t)h * D_ * E_ + (size_t)(kb + kk) * E_ + g * 16;
#pragma unroll
      for (int j = 0; j < 8; ++j) {
        union { unsigned u; _Float16 e[2]; } val;
        val.u = *(const unsigned*)(gw + j * 2);
        int n = g * 16 + j * 2;
        ws[n * 36 + kk]       = val.e[0];
        ws[(n + 1) * 36 + kk] = val.e[1];
      }
    }
#if HAVE_TDM
    if (wid == 0) __builtin_amdgcn_s_wait_tensorcnt(0);  // TDM tile landed
#endif
    __syncthreads();
    v16h a = frag_ld(xs + wid * 16 * 36, 36, 0);
#pragma unroll
    for (int i = 0; i < 4; ++i) {
      v16h bf = frag_ld(ws + i * 16 * 36, 36, 0);
      acc[i]  = wmma16(a, bf, acc[i]);
    }
    __syncthreads();
  }

  // epilogue: + bias, store (V transposed for the P@V WMMA B-operand)
  const int col = lane & 15, hi = lane >> 4;
#pragma unroll
  for (int i = 0; i < 4; ++i) {
    const int n   = i * 16 + col;
    const float bb = bias[h * E_ + n];
#pragma unroll
    for (int r = 0; r < 8; ++r) {
      const int srow = m0 + wid * 16 + r + 8 * hi;
      const float v  = acc[i][r] + bb;
      if (z < 2)
        outp[((size_t)bh * S_ + srow) * E_ + n] = (_Float16)v;
      else
        outp[((size_t)bh * E_ + n) * S_ + srow] = (_Float16)v;
    }
  }
}

// ---------------------------------------------------------------------------
// Flash-style attention: each wave owns a 16-row Q tile, streams K/V in
// 32-column steps with online softmax; ctx accumulated in f32 WMMA accs.
// Block: 256 threads (8 waves). Grid: (S/128, B*H).
// ---------------------------------------------------------------------------
__global__ __launch_bounds__(256) void attn_kernel(
    const _Float16* __restrict__ Q, const _Float16* __restrict__ K,
    const _Float16* __restrict__ Vt, _Float16* __restrict__ ctx) {
  __shared__ __align__(16) _Float16 pt[8 * 16 * 36];  // per-wave P repack tile

  const int bh   = blockIdx.y;
  const int b    = bh / H_, h = bh % H_;
  const int wid  = threadIdx.x >> 5;
  const int lane = threadIdx.x & 31;
  const int m0   = (blockIdx.x * 8 + wid) * 16;

  const _Float16* Qb = Q + (size_t)bh * S_ * E_;
  const _Float16* Kb = K + (size_t)bh * S_ * E_;
  const _Float16* Vb = Vt + (size_t)bh * E_ * S_;
  _Float16*       ptw = pt + wid * 16 * 36;

  const v16h qa0 = frag_ld(Qb + (size_t)m0 * E_, E_, 0);
  const v16h qa1 = frag_ld(Qb + (size_t)m0 * E_, E_, 32);

  float Mr[8], Lr[8];
  v8f acc[4];
#pragma unroll
  for (int r = 0; r < 8; ++r) { Mr[r] = -1e30f; Lr[r] = 0.f; }
#pragma unroll
  for (int i = 0; i < 4; ++i) {
    v8f zv = {0.f, 0.f, 0.f, 0.f, 0.f, 0.f, 0.f, 0.f};
    acc[i] = zv;
  }

  const int col = lane & 15, hi = lane >> 4;

  for (int j = 0; j < S_; j += 32) {
    // scores: two 16x16 tiles, K-dim = E = 64 (2 WMMA steps each)
    v8f s0 = {0.f,0.f,0.f,0.f,0.f,0.f,0.f,0.f};
    v8f s1 = {0.f,0.f,0.f,0.f,0.f,0.f,0.f,0.f};
    const _Float16* kb0 = Kb + (size_t)j * E_;
    const _Float16* kb1 = Kb + (size_t)(j + 16) * E_;
    s0 = wmma16(qa0, frag_ld(kb0, E_, 0), s0);
    s0 = wmma16(qa1, frag_ld(kb0, E_, 32), s0);
    s1 = wmma16(qa0, frag_ld(kb1, E_, 0), s1);
    s1 = wmma16(qa1, frag_ld(kb1, E_, 32), s1);
    if (j + 32 < S_) {
      __builtin_prefetch(Kb + (size_t)(j + 32) * E_, 0, 0);  // global_prefetch_b8
      __builtin_prefetch(Vb + (size_t)(j + 32), 0, 0);
    }

    // online softmax; rows of the C-fragment live in 16-lane halves,
    // so row reductions are xor-shuffles with masks 1,2,4,8 (wave32).
    float sc[8];
#pragma unroll
    for (int r = 0; r < 8; ++r) {
      const float a0 = s0[r] * ISCALE;
      const float a1 = s1[r] * ISCALE;
      float m = fmaxf(a0, a1);
#pragma unroll
      for (int msk = 1; msk < 16; msk <<= 1)
        m = fmaxf(m, __shfl_xor(m, msk, 32));
      const float nm = fmaxf(Mr[r], m);
      sc[r] = __expf(Mr[r] - nm);
      Mr[r] = nm;
      const float p0 = __expf(a0 - nm);
      const float p1 = __expf(a1 - nm);
      float rs = p0 + p1;
#pragma unroll
      for (int msk = 1; msk < 16; msk <<= 1)
        rs += __shfl_xor(rs, msk, 32);
      Lr[r] = Lr[r] * sc[r] + rs;
#pragma unroll
      for (int i = 0; i < 4; ++i) acc[i][r] *= sc[r];
      // spill P (C layout) to per-wave LDS tile for A-layout repack
      const int mrow = r + (hi << 3);
      ptw[mrow * 36 + col]      = (_Float16)p0;
      ptw[mrow * 36 + 16 + col] = (_Float16)p1;
    }
    asm volatile("s_wait_dscnt 0" ::: "memory");   // CDNA5 split DS counter
    const v16h pa = frag_ld(ptw, 36, 0);
    asm volatile("" ::: "memory");

    // ctx += P(16x32) @ V[j:j+32, :]  (V stored transposed: pairs contiguous)
#pragma unroll
    for (int i = 0; i < 4; ++i) {
      v16h vb = frag_ld(Vb + (size_t)(i * 16) * S_ + j, S_, 0);
      acc[i]  = wmma16(pa, vb, acc[i]);
    }
  }

  // normalize and store ctx as [b, s, h, e] (heads concatenated)
#pragma unroll
  for (int r = 0; r < 8; ++r) {
    const float inv = 1.0f / Lr[r];
    const int   s   = m0 + r + 8 * hi;
#pragma unroll
    for (int i = 0; i < 4; ++i) {
      const int e = i * 16 + col;
      ctx[(((size_t)b * S_ + s) * H_ + h) * E_ + e] = (_Float16)(acc[i][r] * inv);
    }
  }
}

// ---------------------------------------------------------------------------
// Output projection + residual + LayerNorm, fused.
// Block: 256 threads (8 waves) -> 16 rows x 768 cols. Grid: B*S/16.
// ---------------------------------------------------------------------------
__global__ __launch_bounds__(256) void proj_ln_kernel(
    const _Float16* __restrict__ ctx, const _Float16* __restrict__ WoT,
    const float* __restrict__ bo, const float* __restrict__ x,
    const float* __restrict__ gamma, const float* __restrict__ beta,
    float* __restrict__ y) {
  __shared__ float hs[16 * D_];
  __shared__ float ps1[256], ps2[256];
  __shared__ float muS[16], rsS[16];

  const int m0   = blockIdx.x * 16;
  const int t    = threadIdx.x;
  const int wid  = t >> 5;
  const int lane = t & 31;

  v8f acc[6];
#pragma unroll
  for (int i = 0; i < 6; ++i) {
    v8f zv = {0.f, 0.f, 0.f, 0.f, 0.f, 0.f, 0.f, 0.f};
    acc[i] = zv;
  }

  const _Float16* arow = ctx + (size_t)m0 * HE_;
  for (int kb = 0; kb < HE_; kb += 32) {
    const v16h a = frag_ld(arow, HE_, kb);
#pragma unroll
    for (int i = 0; i < 6; ++i) {
      const int n0 = wid * 96 + i * 16;
      v16h bf = frag_ld(WoT + (size_t)n0 * HE_, HE_, kb);
      acc[i]  = wmma16(a, bf, acc[i]);
    }
  }

  // h = x + proj + bo  -> LDS (f32) for the row-wise LN reduction
  const int col = lane & 15, hi = lane >> 4;
#pragma unroll
  for (int i = 0; i < 6; ++i) {
    const int n  = wid * 96 + i * 16 + col;
    const float bb = bo[n];
#pragma unroll
    for (int r = 0; r < 8; ++r) {
      const int row = r + 8 * hi;
      hs[row * D_ + n] = acc[i][r] + bb + x[(size_t)(m0 + row) * D_ + n];
    }
  }
  __syncthreads();

  {
    const int row = t >> 4, seg = t & 15;
    float s1 = 0.f, s2 = 0.f;
    for (int i = 0; i < 48; ++i) {
      const float v = hs[row * D_ + seg * 48 + i];
      s1 += v;
      s2 += v * v;
    }
    ps1[t] = s1;
    ps2[t] = s2;
  }
  __syncthreads();
  if (t < 16) {
    float s1 = 0.f, s2 = 0.f;
    for (int i = 0; i < 16; ++i) { s1 += ps1[t * 16 + i]; s2 += ps2[t * 16 + i]; }
    const float mu  = s1 / (float)D_;
    const float var = s2 / (float)D_ - mu * mu;
    muS[t] = mu;
    rsS[t] = rsqrtf(var + LN_EPS);
  }
  __syncthreads();
  {
    const int row = t >> 4, c0 = (t & 15) * 48;
    const float mu = muS[row], rs = rsS[row];
    for (int i = 0; i < 48; ++i) {
      const int n = c0 + i;
      y[(size_t)(m0 + row) * D_ + n] = (hs[row * D_ + n] - mu) * rs * gamma[n] + beta[n];
    }
  }
}

// ---------------------------------------------------------------------------
// Host launcher
// ---------------------------------------------------------------------------
extern "C" void kernel_launch(void* const* d_in, const int* in_sizes, int n_in,
                              void* d_out, int out_size, void* d_ws, size_t ws_size,
                              hipStream_t stream) {
  (void)in_sizes; (void)n_in; (void)out_size; (void)ws_size;
  const float* x     = (const float*)d_in[0];
  const float* Wq    = (const float*)d_in[1];
  const float* bq    = (const float*)d_in[2];
  const float* Wk    = (const float*)d_in[3];
  const float* bk    = (const float*)d_in[4];
  const float* Wv    = (const float*)d_in[5];
  const float* bv    = (const float*)d_in[6];
  const float* Wo    = (const float*)d_in[7];
  const float* bo    = (const float*)d_in[8];
  const float* gamma = (const float*)d_in[9];
  const float* beta  = (const float*)d_in[10];
  float* y = (float*)d_out;

  char* ws   = (char*)d_ws;
  size_t off = 0;
  auto alloc = [&](size_t bytes) {
    char* p = ws + off;
    off += (bytes + 255) & ~(size_t)255;
    return p;
  };
  _Float16* xh  = (_Float16*)alloc((size_t)B_ * S_ * D_ * 2);
  _Float16* Wqh = (_Float16*)alloc((size_t)H_ * D_ * E_ * 2);
  _Float16* Wkh = (_Float16*)alloc((size_t)H_ * D_ * E_ * 2);
  _Float16* Wvh = (_Float16*)alloc((size_t)H_ * D_ * E_ * 2);
  _Float16* WoT = (_Float16*)alloc((size_t)HE_ * D_ * 2);
  _Float16* Qf  = (_Float16*)alloc((size_t)B_ * H_ * S_ * E_ * 2);
  _Float16* Kf  = (_Float16*)alloc((size_t)B_ * H_ * S_ * E_ * 2);
  _Float16* Vt  = (_Float16*)alloc((size_t)B_ * H_ * S_ * E_ * 2);
  _Float16* Ctx = (_Float16*)alloc((size_t)B_ * S_ * H_ * E_ * 2);

  const int nx = B_ * S_ * D_;
  cast_f16_kernel<<<(nx + 255) / 256, 256, 0, stream>>>(x, xh, nx);
  const int nw = H_ * D_ * E_;
  cast_f16_kernel<<<(nw + 255) / 256, 256, 0, stream>>>(Wq, Wqh, nw);
  cast_f16_kernel<<<(nw + 255) / 256, 256, 0, stream>>>(Wk, Wkh, nw);
  cast_f16_kernel<<<(nw + 255) / 256, 256, 0, stream>>>(Wv, Wvh, nw);
  packT_kernel<<<dim3(D_ / 16, HE_ / 16), dim3(16, 16), 0, stream>>>(Wo, WoT, HE_, D_);

  qkv_kernel<<<dim3(S_ / 64, B_ * H_, 3), 128, 0, stream>>>(
      xh, Wqh, Wkh, Wvh, bq, bk, bv, Qf, Kf, Vt);

  attn_kernel<<<dim3(S_ / 128, B_ * H_), 256, 0, stream>>>(Qf, Kf, Vt, Ctx);

  proj_ln_kernel<<<dim3(B_ * S_ / 16), 256, 0, stream>>>(
      Ctx, WoT, bo, x, gamma, beta, y);
}
